// ConvAttention_15985868276215
// MI455X (gfx1250) — matrix operand order repaired and checked by reading, not compiled, not run on hardware
//
#include <hip/hip_runtime.h>
#include <hip/hip_bf16.h>

// ---------------------------------------------------------------------------
// Spiking conv-attention for MI455X (gfx1250, wave32, WMMA).
//   q,k,v = sepconv(x)        (fp32 direct conv + BN, 6.7 GF)
//   dots  = q·kT * 50^-0.5    (f16 WMMA, f32 accum)
//   attn  = spiking softmax   (parallel row softmax + column prefix-scan)
//   out   = attn·v            (f16 WMMA, LDS-staged B, prefetched A stream)
// ---------------------------------------------------------------------------

typedef _Float16 h8  __attribute__((ext_vector_type(8)));
typedef _Float16 h16 __attribute__((ext_vector_type(16)));
typedef float    f8  __attribute__((ext_vector_type(8)));

#define NB    4       // batch
#define NC    50      // channels / head dim
#define NIMG  64
#define NN    4096    // tokens = 64*64
#define KD    64      // head dim padded to 64 for K-tiling (zeros beyond 50)
#define P1    66      // conv1 padded size
#define O1    62      // conv1 out size
#define P2    65      // conv2 padded size
#define DTQ   0.001f
#define BNEPS 1e-5f
#define SCALE 0.14142135623730951f   // 50^-0.5
#define LROW  40      // LDS row stride (halves): bank stride 20 -> conflict-free

__device__ __forceinline__ h16 cat16(h8 lo, h8 hi) {
  return __builtin_shufflevector(lo, hi, 0,1,2,3,4,5,6,7,8,9,10,11,12,13,14,15);
}

// ---- K1: (B, 4096, 50) -> zero-padded NCHW image xi (B, 50, 66, 66) --------
__global__ void k1_rearrange(const float* __restrict__ x, float* __restrict__ xi) {
  int idx = blockIdx.x * blockDim.x + threadIdx.x;
  int total = NB * NC * P1 * P1;
  if (idx >= total) return;
  int w = idx % P1;
  int h = (idx / P1) % P1;
  int c = (idx / (P1 * P1)) % NC;
  int b = idx / (P1 * P1 * NC);
  int hr = h - 2;                       // pad top=2
  float v = 0.f;
  if (hr >= 0 && hr < NIMG && w < NIMG) // pad right=2
    v = x[((size_t)b * NN + hr * NIMG + w) * NC + c];
  xi[idx] = v;
}

// ---- K2: conv 5x5 (50->50) + eval BatchNorm, into zero-padded 65x65 --------
__global__ void k2_conv1_bn(const float* __restrict__ xi, const float* __restrict__ w1,
                            const float* __restrict__ gam, const float* __restrict__ bet,
                            const float* __restrict__ mu,  const float* __restrict__ var,
                            float* __restrict__ y1p) {
  int idx = blockIdx.x * blockDim.x + threadIdx.x;
  int total = 3 * NB * NC * O1 * O1;
  if (idx >= total) return;
  int ow = idx % O1;
  int oh = (idx / O1) % O1;
  int co = (idx / (O1 * O1)) % NC;
  int b  = (idx / (O1 * O1 * NC)) % NB;
  int i  = idx / (O1 * O1 * NC * NB);
  float acc = 0.f;
  for (int ci = 0; ci < NC; ++ci) {
    const float* xp = xi + ((size_t)(b * NC + ci) * P1 + oh) * P1 + ow;
    const float* wp = w1 + ((size_t)((i * NC + co) * NC + ci)) * 25;
#pragma unroll
    for (int kh = 0; kh < 5; ++kh)
#pragma unroll
      for (int kw = 0; kw < 5; ++kw)
        acc = fmaf(xp[kh * P1 + kw], wp[kh * 5 + kw], acc);
  }
  int pc = i * NC + co;
  float inv = rsqrtf(var[pc] + BNEPS);
  float v = (acc - mu[pc]) * (gam[pc] * inv) + bet[pc];
  // pad top=3, right=3 -> write at row oh+3, col ow of a zeroed 65x65 plane
  y1p[((size_t)(i * NB + b) * NC + co) * (P2 * P2) + (oh + 3) * P2 + ow] = v;
}

// ---- K3: conv 2x2 (50->50), pack to WMMA-friendly f16 layouts --------------
// q -> qh [B][4096][64] row-major (A operand; K padded w/ zeros)
// k -> kh [B][4096][64] row-major (B operand of GEMM1: per-lane contiguous K)
// v -> vT [B][64][4096]           (B operand of GEMM2: per-lane contiguous K)
__global__ void k3_conv2_pack(const float* __restrict__ y1p, const float* __restrict__ w2,
                              _Float16* __restrict__ qh, _Float16* __restrict__ kh,
                              _Float16* __restrict__ vT) {
  int idx = blockIdx.x * blockDim.x + threadIdx.x;   // 3*NB*KD*NN threads
  int n = idx & (NN - 1);
  int d = (idx >> 12) & (KD - 1);
  int b = (idx >> 18) & 3;
  int i = idx >> 20;
  if (i >= 3) return;
  float acc = 0.f;
  if (d < NC) {
    int h = n >> 6, w = n & 63;
    for (int ci = 0; ci < NC; ++ci) {
      const float* yp = y1p + ((size_t)(i * NB + b) * NC + ci) * (P2 * P2) + h * P2 + w;
      const float* wp = w2 + ((size_t)((i * NC + d) * NC + ci)) * 4;
      acc = fmaf(yp[0],      wp[0], acc);
      acc = fmaf(yp[1],      wp[1], acc);
      acc = fmaf(yp[P2],     wp[2], acc);
      acc = fmaf(yp[P2 + 1], wp[3], acc);
    }
  }
  _Float16 hv = (_Float16)acc;
  if (i == 0)      qh[((size_t)b * NN + n) * KD + d] = hv;
  else if (i == 1) kh[((size_t)b * NN + n) * KD + d] = hv;
  else             vT[((size_t)b * KD + d) * NN + n] = hv;
}

// ---- K4: dots = q·kT * scale, f16 WMMA. Wave -> 16 rows x 128 cols ---------
__global__ void k4_gemm_qk(const _Float16* __restrict__ qh, const _Float16* __restrict__ kh,
                           float* __restrict__ dots) {
  int lane = threadIdx.x & 31;
  int wave = threadIdx.x >> 5;
  int b  = blockIdx.z;
  int m0 = blockIdx.y * 16;
  int n0 = blockIdx.x * 512 + wave * 128;
  int lr = lane & 15, hi = lane >> 4;

  // A fragments: rows m0..m0+15, K chunks {0..31},{32..63} of qh (ld = 64)
  const _Float16* ap = qh + ((size_t)b * NN + m0 + lr) * KD;
  h16 a0 = *(const h16*)(ap + 0  + hi * 16);
  h16 a1 = *(const h16*)(ap + 32 + hi * 16);

  f8 acc[8];
#pragma unroll
  for (int t = 0; t < 8; ++t) acc[t] = (f8)(0.f);

#pragma unroll
  for (int t = 0; t < 8; ++t) {
    const _Float16* bp = kh + ((size_t)b * NN + n0 + t * 16 + lr) * KD;
    h16 b0 = *(const h16*)(bp + 0  + hi * 16);
    h16 b1 = *(const h16*)(bp + 32 + hi * 16);
    acc[t] = __builtin_amdgcn_wmma_f32_16x16x32_f16(false, a0, false, b0, (short)0, acc[t], false, false);
    acc[t] = __builtin_amdgcn_wmma_f32_16x16x32_f16(false, a1, false, b1, (short)0, acc[t], false, false);
  }

#pragma unroll
  for (int t = 0; t < 8; ++t)
#pragma unroll
    for (int r = 0; r < 8; ++r)
      dots[((size_t)b * NN + m0 + r + hi * 8) * NN + n0 + t * 16 + lr] = acc[t][r] * SCALE;
}

// ---- K5a: row-wise softmax in place (each row independent) -----------------
__global__ void k5a_softmax_rows(float* __restrict__ dots) {
  __shared__ float red[256];
  int tid = threadIdx.x;
  float* row = dots + ((size_t)blockIdx.y * NN + blockIdx.x) * NN;
  float vals[16];
#pragma unroll
  for (int i = 0; i < 16; ++i) vals[i] = row[tid + i * 256];
  float m = vals[0];
#pragma unroll
  for (int i = 1; i < 16; ++i) m = fmaxf(m, vals[i]);
  red[tid] = m; __syncthreads();
  for (int s = 128; s > 0; s >>= 1) { if (tid < s) red[tid] = fmaxf(red[tid], red[tid + s]); __syncthreads(); }
  m = red[0]; __syncthreads();
  float sum = 0.f;
#pragma unroll
  for (int i = 0; i < 16; ++i) { vals[i] = __expf(vals[i] - m); sum += vals[i]; }
  red[tid] = sum; __syncthreads();
  for (int s = 128; s > 0; s >>= 1) { if (tid < s) red[tid] += red[tid + s]; __syncthreads(); }
  float inv = 1.f / red[0];
#pragma unroll
  for (int i = 0; i < 16; ++i) row[tid + i * 256] = vals[i] * inv;
}

// ---- K5b: per-column partial sums of r*dt over 128-row chunks --------------
__global__ void k5b_colsum(const float* __restrict__ r, float* __restrict__ S) {
  int col   = blockIdx.x * 256 + threadIdx.x;
  int chunk = blockIdx.y;
  int b     = blockIdx.z;
  const float* base = r + (size_t)b * NN * NN + (size_t)chunk * 128 * NN + col;
  float s = 0.f;
  for (int t = 0; t < 128; ++t) s += base[(size_t)t * NN] * DTQ;
  S[((size_t)b * 32 + chunk) * NN + col] = s;
}

// ---- K5c: exclusive scan of the 32 chunk sums per column (in place) --------
__global__ void k5c_scan(float* __restrict__ S) {
  int col = blockIdx.x * 256 + threadIdx.x;
  int b   = blockIdx.y;
  float run = 0.f;
  for (int c = 0; c < 32; ++c) {
    size_t idx = ((size_t)b * 32 + c) * NN + col;
    float t = S[idx];
    S[idx] = run;
    run += t;
  }
}

// ---- K5d: emit spikes  s_t = floor(C_t) - floor(C_{t-1});  attn = s/dt -----
__global__ void k5d_emit(const float* __restrict__ r, const float* __restrict__ S,
                         _Float16* __restrict__ attn) {
  int col   = blockIdx.x * 256 + threadIdx.x;
  int chunk = blockIdx.y;
  int b     = blockIdx.z;
  float c  = S[((size_t)b * 32 + chunk) * NN + col];
  float fc = floorf(c);
  const float* rb = r    + (size_t)b * NN * NN + (size_t)chunk * 128 * NN + col;
  _Float16*    ab = attn + (size_t)b * NN * NN + (size_t)chunk * 128 * NN + col;
  for (int t = 0; t < 128; ++t) {
    float cn = c + rb[(size_t)t * NN] * DTQ;
    float fn = floorf(cn);
    ab[(size_t)t * NN] = (_Float16)((fn - fc) * 1000.0f);   // s/dt
    c = cn; fc = fn;
  }
}

// ---- K6: out = attn·v, f16 WMMA, K = 4096 ----------------------------------
// 4 waves / block share the B operand: each 64x32 vT K-chunk is staged once in
// LDS (double-buffered, conflict-free 40-half row stride), B fragments come
// from ds_load_b128. The streaming attn (A) rows are prefetched 4 chunks ahead
// (global_prefetch_b8).
__global__ void k6_gemm_av(const _Float16* __restrict__ attn, const _Float16* __restrict__ vT,
                           float* __restrict__ out) {
  __shared__ __align__(32) _Float16 bs[2][KD * LROW];
  int tid  = threadIdx.x;
  int lane = tid & 31;
  int wave = tid >> 5;
  int b  = blockIdx.y;
  int m0 = (blockIdx.x * 4 + wave) * 16;
  int lr = lane & 15, hi = lane >> 4;

  // staging role: thread -> (row d, 16-half half) of the 64x32 chunk
  int sd = tid >> 1, sh = tid & 1;
  const _Float16* vrow = vT + ((size_t)b * KD + sd) * NN + sh * 16;

  // preload chunk 0 into LDS buffer 0
  h8 st0 = *(const h8*)(vrow);
  h8 st1 = *(const h8*)(vrow + 8);
  *(h8*)&bs[0][sd * LROW + sh * 16]     = st0;
  *(h8*)&bs[0][sd * LROW + sh * 16 + 8] = st1;

  f8 acc[4];
#pragma unroll
  for (int t = 0; t < 4; ++t) acc[t] = (f8)(0.f);

  const _Float16* arow = attn + ((size_t)b * NN + m0 + lr) * NN;
  __syncthreads();

  for (int kc = 0; kc < NN / 32; ++kc) {
    int cur = kc & 1;
    // stage next chunk into registers while this chunk computes
    if (kc + 1 < NN / 32) {
      st0 = *(const h8*)(vrow + (kc + 1) * 32);
      st1 = *(const h8*)(vrow + (kc + 1) * 32 + 8);
    }
    const _Float16* ap = arow + kc * 32;
    __builtin_prefetch(ap + 128, 0, 0);          // attn stream, 4 chunks ahead
    h16 a = *(const h16*)(ap + hi * 16);
#pragma unroll
    for (int t = 0; t < 4; ++t) {
      int o = (t * 16 + lr) * LROW + hi * 16;
      h16 bf = cat16(*(const h8*)&bs[cur][o], *(const h8*)&bs[cur][o + 8]);
      acc[t] = __builtin_amdgcn_wmma_f32_16x16x32_f16(false, a, false, bf, (short)0, acc[t], false, false);
    }
    __syncthreads();
    if (kc + 1 < NN / 32) {
      *(h8*)&bs[1 - cur][sd * LROW + sh * 16]     = st0;
      *(h8*)&bs[1 - cur][sd * LROW + sh * 16 + 8] = st1;
      __syncthreads();
    }
  }

#pragma unroll
  for (int t = 0; t < 4; ++t) {
#pragma unroll
    for (int r = 0; r < 8; ++r) {
      int dcol = t * 16 + lr;
      if (dcol < NC)
        out[((size_t)b * NN + m0 + r + hi * 8) * NC + dcol] = acc[t][r];
    }
  }
}

// ---------------------------------------------------------------------------
extern "C" void kernel_launch(void* const* d_in, const int* in_sizes, int n_in,
                              void* d_out, int out_size, void* d_ws, size_t ws_size,
                              hipStream_t stream) {
  (void)in_sizes; (void)n_in; (void)out_size; (void)ws_size;
  const float* x   = (const float*)d_in[0];
  const float* w1  = (const float*)d_in[1];
  const float* w2  = (const float*)d_in[2];
  const float* gam = (const float*)d_in[3];
  const float* bet = (const float*)d_in[4];
  const float* mu  = (const float*)d_in[5];
  const float* var = (const float*)d_in[6];

  char* ws = (char*)d_ws;
  size_t off = 0;
  auto alloc = [&](size_t bytes) { char* p = ws + off; off += (bytes + 255) & ~(size_t)255; return p; };

  float*    xi   = (float*)   alloc((size_t)NB * NC * P1 * P1 * 4);
  float*    y1p  = (float*)   alloc((size_t)3 * NB * NC * P2 * P2 * 4);
  _Float16* qh   = (_Float16*)alloc((size_t)NB * NN * KD * 2);
  _Float16* kh   = (_Float16*)alloc((size_t)NB * NN * KD * 2);
  _Float16* vT   = (_Float16*)alloc((size_t)NB * KD * NN * 2);
  float*    dots = (float*)   alloc((size_t)NB * NN * NN * 4);
  _Float16* attn = (_Float16*)alloc((size_t)NB * NN * NN * 2);
  float*    S    = (float*)   alloc((size_t)NB * 32 * NN * 4);

  hipMemsetAsync(y1p, 0, (size_t)3 * NB * NC * P2 * P2 * 4, stream);

  {
    int total = NB * NC * P1 * P1;
    k1_rearrange<<<(total + 255) / 256, 256, 0, stream>>>(x, xi);
  }
  {
    int total = 3 * NB * NC * O1 * O1;
    k2_conv1_bn<<<(total + 255) / 256, 256, 0, stream>>>(xi, w1, gam, bet, mu, var, y1p);
  }
  {
    int total = 3 * NB * KD * NN;
    k3_conv2_pack<<<(total + 255) / 256, 256, 0, stream>>>(y1p, w2, qh, kh, vT);
  }
  k4_gemm_qk<<<dim3(8, NN / 16, NB), 128, 0, stream>>>(qh, kh, dots);
  k5a_softmax_rows<<<dim3(NN, NB), 256, 0, stream>>>(dots);
  k5b_colsum<<<dim3(NN / 256, 32, NB), 256, 0, stream>>>(dots, S);
  k5c_scan<<<dim3(NN / 256, NB), 256, 0, stream>>>(S);
  k5d_emit<<<dim3(NN / 256, 32, NB), 256, 0, stream>>>(dots, S, attn);
  k6_gemm_av<<<dim3(NN / 64, NB), 128, 0, stream>>>(attn, vT, (float*)d_out);
}
